// LocalTransformerDecoderLayer_73787538145798
// MI455X (gfx1250) — matrix operand order repaired and checked by reading, not compiled, not run on hardware
//
#include <hip/hip_runtime.h>
#include <hip/hip_bf16.h>

#define BB 8
#define C 256
#define NQ 1024
#define NK 4096
#define S 16
#define H 8
#define FF 2048
#define DH 32
#define BQ (BB * NQ)
#define KVP 264    // padded LDS row stride (bf16 elems) for C=256 tiles
#define AP2 2056   // padded LDS row stride for FF=2048 tiles

typedef __attribute__((ext_vector_type(16))) __bf16 v16bf;
typedef __attribute__((ext_vector_type(8)))  __bf16 v8bf;
typedef __attribute__((ext_vector_type(8)))  float  v8f;

#define WMMA_BF16(a, b, c) \
  __builtin_amdgcn_wmma_f32_16x16x32_bf16(false, (a), false, (b), (short)0, (c), false, false)

// A fragment (16x32 bf16, M x K) from an LDS row-major tile with padded stride.
// lanes 0-15: e0-15 -> K=kk+0..15
// lanes 16-31: e0-15 -> K=kk+16..31
__device__ __forceinline__ v16bf load_a_lds(const __bf16* base, int lane, int kk, int stride) {
  int m = lane & 15;
  int off = (lane & 16) ? 16 : 0;
  const __bf16* p = base + m * stride + kk + off;
  v8bf lo = *(const v8bf*)p;
  v8bf hi = *(const v8bf*)(p + 8);
  v16bf a;
#pragma unroll
  for (int e = 0; e < 8; ++e) { a[e] = lo[e]; a[8 + e] = hi[e]; }
  return a;
}

// B fragment (32x16 bf16, K x N) for y = x @ W^T: B[k][n] = W[n][k].
// lane holds column n = n0 + (lane&15); K = kk + (lane<16 ? 0..15 : 16..31), contiguous in W row.
__device__ __forceinline__ v16bf load_b_glb(const __bf16* w, int lane, int n0, int kk, int ldb) {
  const __bf16* p = w + (size_t)(n0 + (lane & 15)) * ldb + kk + ((lane & 16) ? 16 : 0);
  return *(const v16bf*)p;
}

// 256xC tile (2 m-tiles per wave) x W^T -> sDst, with B-fragment reuse across both
// m-tiles (two independent back-to-back WMMAs per weight load).
__device__ __forceinline__ void kv_proj_pair(const __bf16* __restrict__ sSrc,
                                             __bf16* __restrict__ sDst,
                                             const __bf16* __restrict__ w,
                                             const float* __restrict__ bv,
                                             int lane, int wave) {
  const __bf16* ab0 = sSrc + (wave * 2 + 0) * 16 * KVP;
  const __bf16* ab1 = sSrc + (wave * 2 + 1) * 16 * KVP;
  v16bf a0[8], a1[8];
#pragma unroll
  for (int k8 = 0; k8 < 8; ++k8) {
    a0[k8] = load_a_lds(ab0, lane, k8 * 32, KVP);
    a1[k8] = load_a_lds(ab1, lane, k8 * 32, KVP);
  }
  int nl = lane & 15;
  int mo = (lane & 16) ? 8 : 0;
  int row0 = (wave * 2 + 0) * 16 + mo;
  int row1 = (wave * 2 + 1) * 16 + mo;
  for (int nt = 0; nt < 16; ++nt) {
    int n0 = nt * 16;
    float bias = bv[n0 + nl];
    v8f acc0, acc1;
#pragma unroll
    for (int r = 0; r < 8; ++r) { acc0[r] = bias; acc1[r] = bias; }
#pragma unroll
    for (int k8 = 0; k8 < 8; ++k8) {
      v16bf bm = load_b_glb(w, lane, n0, k8 * 32, C);
      acc0 = WMMA_BF16(a0[k8], bm, acc0);
      acc1 = WMMA_BF16(a1[k8], bm, acc1);
    }
#pragma unroll
    for (int r = 0; r < 8; ++r) {
      sDst[(row0 + r) * KVP + n0 + nl] = (__bf16)acc0[r];
      sDst[(row1 + r) * KVP + n0 + nl] = (__bf16)acc1[r];
    }
  }
}

// ---------------------------------------------------------------- weights -> bf16
__global__ __launch_bounds__(256) void cvt_bf16_kernel(const float* __restrict__ src,
                                                       __bf16* __restrict__ dst, int n) {
  int i = blockIdx.x * 256 + threadIdx.x;
  if (i < n) dst[i] = (__bf16)src[i];
}

// ---------------------------------------------------------------- key (B,C,NK) -> keyT (B,NK,C) bf16
__global__ __launch_bounds__(256) void transpose_key_kernel(const float* __restrict__ key,
                                                            __bf16* __restrict__ keyT) {
  __shared__ float tile[32][33];
  int k0 = blockIdx.x * 32, c0 = blockIdx.y * 32, b = blockIdx.z;
  int tx = threadIdx.x, ty = threadIdx.y;  // block (32,8)
#pragma unroll
  for (int j = 0; j < 4; ++j)
    tile[ty + j * 8][tx] = key[((size_t)b * C + c0 + ty + j * 8) * NK + k0 + tx];
  __syncthreads();
#pragma unroll
  for (int j = 0; j < 4; ++j)
    keyT[((size_t)b * NK + k0 + ty + j * 8) * C + c0 + tx] = (__bf16)tile[tx][ty + j * 8];
}

// ---------------------------------------------------------------- box query: first S inside keys
__global__ __launch_bounds__(256) void box_query_kernel(const float* __restrict__ query_pos,
                                                        const float* __restrict__ key_pos,
                                                        int* __restrict__ idx,
                                                        unsigned* __restrict__ maskb) {
  __shared__ float kp[NK * 3];
  int b = blockIdx.x >> 2;  // NQ/256 = 4 blocks per batch
  int tid = threadIdx.x;
  for (int i = tid; i < NK * 3; i += 256) kp[i] = key_pos[(size_t)b * NK * 3 + i];
  __syncthreads();
  int q = ((blockIdx.x & 3) << 8) + tid;
  int bq = b * NQ + q;
  const float* qp = query_pos + (size_t)bq * 6;
  float cx = qp[0], cy = qp[1], cz = qp[2];
  float hx = 0.5f * qp[3], hy = 0.5f * qp[4], hz = 0.5f * qp[5];
  int id[S];
  int cnt = 0;
  for (int k = 0; k < NK; ++k) {
    float dx = fabsf(kp[3 * k + 0] - cx);
    float dy = fabsf(kp[3 * k + 1] - cy);
    float dz = fabsf(kp[3 * k + 2] - cz);
    if (dx <= hx && dy <= hy && dz <= hz) {
      if (cnt < S) id[cnt] = k;
      ++cnt;
      if (cnt == S) break;
    }
  }
  unsigned mb = 0;
#pragma unroll
  for (int s = 0; s < S; ++s) {
    int v = (s < cnt) ? id[s] : 0;
    if (s >= cnt) mb |= (1u << s);
    idx[(size_t)bq * S + s] = v;
  }
  maskb[bq] = mb & ~1u;  // slot 0 always attends
}

// ---------------------------------------------------------------- qtok, q_in=qtok+qpe, gxyz
__global__ __launch_bounds__(256) void build_kernel(const float* __restrict__ query,
                                                    const float* __restrict__ query_pos,
                                                    const float* __restrict__ key_pos,
                                                    const float* __restrict__ qpe_w,
                                                    const float* __restrict__ qpe_b,
                                                    const int* __restrict__ idx,
                                                    __bf16* __restrict__ qtok,
                                                    __bf16* __restrict__ qin,
                                                    float* __restrict__ gxyz) {
  int bq = blockIdx.x;
  int b = bq / NQ, q = bq % NQ;
  int c = threadIdx.x;
  const float* qp = query_pos + (size_t)bq * 6;
  float qv = query[((size_t)b * C + c) * NQ + q];
  float pe = qpe_b[c];
#pragma unroll
  for (int d = 0; d < 6; ++d) pe += qp[d] * qpe_w[c * 6 + d];
  qtok[(size_t)bq * C + c] = (__bf16)qv;
  qin[(size_t)bq * C + c] = (__bf16)(qv + pe);
  if (c < S * 3) {
    int s = c / 3, d = c % 3;
    int k = idx[(size_t)bq * S + s];
    gxyz[(size_t)bq * S * 3 + c] = key_pos[((size_t)b * NK + k) * 3 + d] - qp[d];
  }
}

// ---------------------------------------------------------------- fused gather+kpe+QKV+attention
__global__ __launch_bounds__(256) void attn_kernel(
    const __bf16* __restrict__ keyT, const __bf16* __restrict__ qin,
    const float* __restrict__ gxyz, const int* __restrict__ idx,
    const unsigned* __restrict__ maskb,
    const float* __restrict__ kpe_w, const float* __restrict__ kpe_b,
    const __bf16* __restrict__ wq, const float* __restrict__ q_b,
    const __bf16* __restrict__ wk, const float* __restrict__ k_b,
    const __bf16* __restrict__ wv, const float* __restrict__ v_b,
    __bf16* __restrict__ ao) {
  extern __shared__ __align__(16) char smem[];
  __bf16* sKV = (__bf16*)smem;          // 256 x KVP (kv_in tile)            135168 B
  __bf16* sKH = sKV + 256 * KVP;        // 256 x KVP (kh, then vh)           135168 B
  __bf16* sQ = sKH + 256 * KVP;         // 16 x KVP  (q_in tile)               8448 B
  float* sQH = (float*)(sQ + 16 * KVP); // 16 x 256  (qh f32)                 16384 B
  float* sG = sQH + 16 * 256;           // 768 gxyz                            3072 B
  int* sIdx = (int*)(sG + 16 * S * 3);  // 256 indices                         1024 B

  int tid = threadIdx.x;
  int lane = tid & 31, wave = tid >> 5;
  int bq0 = blockIdx.x * 16;
  int b = bq0 / NQ;

  // stage q tile, indices, gxyz
  for (int i = tid; i < 16 * C; i += 256)
    sQ[(i >> 8) * KVP + (i & 255)] = qin[(size_t)bq0 * C + i];
  sIdx[tid] = idx[(size_t)bq0 * S + tid];  // 16 q * 16 s = 256
  for (int i = tid; i < 16 * S * 3; i += 256)
    sG[i] = gxyz[(size_t)bq0 * S * 3 + i];
  __syncthreads();

  // build kv_in tile: gather keyT rows + kpe (3 -> C linear), thread = column
  {
    int c = tid;
    float k0 = kpe_w[c * 3 + 0], k1 = kpe_w[c * 3 + 1], k2 = kpe_w[c * 3 + 2];
    float kb = kpe_b[c];
    for (int r = 0; r < 256; ++r) {
      int ki = sIdx[r];
      float v = (float)keyT[((size_t)b * NK + ki) * C + c];
      v += sG[r * 3] * k0 + sG[r * 3 + 1] * k1 + sG[r * 3 + 2] * k2 + kb;
      sKV[r * KVP + c] = (__bf16)v;
    }
  }
  __syncthreads();

  // qh = q_tile @ wq^T + q_b: A fragments hoisted, reused across this wave's 2 n-tiles
  {
    v16bf aq[8];
#pragma unroll
    for (int k8 = 0; k8 < 8; ++k8) aq[k8] = load_a_lds(sQ, lane, k8 * 32, KVP);
    for (int t = 0; t < 2; ++t) {
      int n0 = (wave * 2 + t) * 16;
      float bias = q_b[n0 + (lane & 15)];
      v8f acc;
#pragma unroll
      for (int r = 0; r < 8; ++r) acc[r] = bias;
#pragma unroll
      for (int k8 = 0; k8 < 8; ++k8) {
        v16bf bm = load_b_glb(wq, lane, n0, k8 * 32, C);
        acc = WMMA_BF16(aq[k8], bm, acc);
      }
      int n = n0 + (lane & 15);
      int mo = (lane & 16) ? 8 : 0;
#pragma unroll
      for (int r = 0; r < 8; ++r) sQH[(r + mo) * 256 + n] = acc[r];
    }
  }

  // kh = kv_tile @ wk^T + k_b (A and B fragments register-reused)
  kv_proj_pair(sKV, sKH, wk, k_b, lane, wave);
  __syncthreads();

  // scores + softmax over S=16 keys: thread (qi,h), 128 active
  float att[S];
  int qi = tid >> 3, h = tid & 7;
  if (tid < 128) {
    unsigned mb = maskb[bq0 + qi];
    const float* qh = sQH + qi * 256 + h * DH;
    float mx = -3.0e38f;
#pragma unroll
    for (int s = 0; s < S; ++s) {
      float dot = 0.f;
      const __bf16* kr = sKH + (qi * S + s) * KVP + h * DH;
#pragma unroll
      for (int d = 0; d < DH; ++d) dot += qh[d] * (float)kr[d];
      dot *= 0.17677669529663687f;  // 1/sqrt(32)
      if ((mb >> s) & 1u) dot = -1.0e30f;
      att[s] = dot;
      mx = fmaxf(mx, dot);
    }
    float sum = 0.f;
#pragma unroll
    for (int s = 0; s < S; ++s) { att[s] = __expf(att[s] - mx); sum += att[s]; }
    float inv = 1.0f / sum;
#pragma unroll
    for (int s = 0; s < S; ++s) att[s] *= inv;
  }
  __syncthreads();

  // vh = kv_tile @ wv^T + v_b, overwrite sKH
  kv_proj_pair(sKV, sKH, wv, v_b, lane, wave);
  __syncthreads();

  // ao = attn @ vh
  if (tid < 128) {
#pragma unroll 4
    for (int d = 0; d < DH; ++d) {
      float acc = 0.f;
#pragma unroll
      for (int s = 0; s < S; ++s)
        acc += att[s] * (float)sKH[(qi * S + s) * KVP + h * DH + d];
      ao[(size_t)(bq0 + qi) * C + h * DH + d] = (__bf16)acc;
    }
  }
}

// ---------------------------------------------------------------- row-wise LN over 16x256 LDS tile
__device__ __forceinline__ void ln_16x256(float* sY, const float* g, const float* bt,
                                          float* red, float* stat) {
  int tid = threadIdx.x;
  int q = tid >> 4, j = tid & 15;
  float s = 0.f;
#pragma unroll
  for (int i = 0; i < 16; ++i) s += sY[q * 256 + j + (i << 4)];
  red[q * 16 + j] = s;
  __syncthreads();
  if (tid < 16) {
    float m = 0.f;
#pragma unroll
    for (int i = 0; i < 16; ++i) m += red[tid * 16 + i];
    stat[tid] = m * (1.f / 256.f);
  }
  __syncthreads();
  float mean = stat[q];
  float v = 0.f;
#pragma unroll
  for (int i = 0; i < 16; ++i) {
    float d = sY[q * 256 + j + (i << 4)] - mean;
    v += d * d;
  }
  red[q * 16 + j] = v;
  __syncthreads();
  if (tid < 16) {
    float m = 0.f;
#pragma unroll
    for (int i = 0; i < 16; ++i) m += red[tid * 16 + i];
    stat[16 + tid] = m * (1.f / 256.f);
  }
  __syncthreads();
  float rstd = rsqrtf(stat[16 + q] + 1e-5f);
#pragma unroll
  for (int i = 0; i < 16; ++i) {
    int cc = j + (i << 4);
    sY[q * 256 + cc] = (sY[q * 256 + cc] - mean) * rstd * g[cc] + bt[cc];
  }
  __syncthreads();
}

// ---------------------------------------------------------------- o-proj + residual + LN1
__global__ __launch_bounds__(256) void oproj_ln1_kernel(
    const __bf16* __restrict__ ao, const __bf16* __restrict__ wo,
    const float* __restrict__ o_b, const __bf16* __restrict__ qtok,
    const float* __restrict__ n1_g, const float* __restrict__ n1_b,
    __bf16* __restrict__ x1) {
  extern __shared__ __align__(16) char smem[];
  __bf16* sA = (__bf16*)smem;           // 16 x KVP
  float* sY = (float*)(sA + 16 * KVP);  // 16 x 256
  __shared__ float red[256];
  __shared__ float stat[32];
  int tid = threadIdx.x, lane = tid & 31, wave = tid >> 5;
  int bq0 = blockIdx.x * 16;
  for (int i = tid; i < 16 * C; i += 256)
    sA[(i >> 8) * KVP + (i & 255)] = ao[(size_t)bq0 * C + i];
  __syncthreads();
  {
    v16bf a[8];
#pragma unroll
    for (int k8 = 0; k8 < 8; ++k8) a[k8] = load_a_lds(sA, lane, k8 * 32, KVP);
    for (int t = 0; t < 2; ++t) {
      int n0 = (wave * 2 + t) * 16;
      float bias = o_b[n0 + (lane & 15)];
      v8f acc;
#pragma unroll
      for (int r = 0; r < 8; ++r) acc[r] = bias;
#pragma unroll
      for (int k8 = 0; k8 < 8; ++k8) {
        v16bf bm = load_b_glb(wo, lane, n0, k8 * 32, C);
        acc = WMMA_BF16(a[k8], bm, acc);
      }
      int n = n0 + (lane & 15);
      int mo = (lane & 16) ? 8 : 0;
#pragma unroll
      for (int r = 0; r < 8; ++r) {
        int m = r + mo;
        sY[m * 256 + n] = acc[r] + (float)qtok[(size_t)(bq0 + m) * C + n];
      }
    }
  }
  __syncthreads();
  ln_16x256(sY, n1_g, n1_b, red, stat);
  for (int i = tid; i < 16 * C; i += 256)
    x1[(size_t)bq0 * C + i] = (__bf16)sY[(i >> 8) * 256 + (i & 255)];
}

// ---------------------------------------------------------------- FFN layer 1 (+relu)
__global__ __launch_bounds__(256) void ffn1_kernel(const __bf16* __restrict__ x1,
                                                   const __bf16* __restrict__ w1,
                                                   const float* __restrict__ b1,
                                                   __bf16* __restrict__ hbuf) {
  extern __shared__ __align__(16) char smem[];
  __bf16* sA = (__bf16*)smem;  // 16 x KVP
  int tid = threadIdx.x, lane = tid & 31, wave = tid >> 5;
  int bq0 = blockIdx.x * 16;
  int nbase = blockIdx.y * 256;
  for (int i = tid; i < 16 * C; i += 256)
    sA[(i >> 8) * KVP + (i & 255)] = x1[(size_t)bq0 * C + i];
  __syncthreads();
  {
    v16bf a[8];
#pragma unroll
    for (int k8 = 0; k8 < 8; ++k8) a[k8] = load_a_lds(sA, lane, k8 * 32, KVP);
    for (int t = 0; t < 2; ++t) {
      int n0 = nbase + (wave * 2 + t) * 16;
      float bias = b1[n0 + (lane & 15)];
      v8f acc;
#pragma unroll
      for (int r = 0; r < 8; ++r) acc[r] = bias;
#pragma unroll
      for (int k8 = 0; k8 < 8; ++k8) {
        v16bf bm = load_b_glb(w1, lane, n0, k8 * 32, C);
        acc = WMMA_BF16(a[k8], bm, acc);
      }
      int n = n0 + (lane & 15);
      int mo = (lane & 16) ? 8 : 0;
#pragma unroll
      for (int r = 0; r < 8; ++r)
        hbuf[(size_t)(bq0 + r + mo) * FF + n] = (__bf16)fmaxf(acc[r], 0.f);
    }
  }
}

// ---------------------------------------------------------------- FFN layer 2 + residual + LN2 + transposed out
__global__ __launch_bounds__(256) void ffn2_ln2_out_kernel(
    const __bf16* __restrict__ hbuf, const __bf16* __restrict__ w2,
    const float* __restrict__ b2, const __bf16* __restrict__ x1,
    const float* __restrict__ n2_g, const float* __restrict__ n2_b,
    float* __restrict__ out) {
  extern __shared__ __align__(16) char smem[];
  __bf16* sA = (__bf16*)smem;           // 16 x AP2
  float* sY = (float*)(sA + 16 * AP2);  // 16 x 256
  __shared__ float red[256];
  __shared__ float stat[32];
  int tid = threadIdx.x, lane = tid & 31, wave = tid >> 5;
  int bq0 = blockIdx.x * 16;
  for (int i = tid; i < 16 * FF; i += 256)
    sA[(i >> 11) * AP2 + (i & 2047)] = hbuf[(size_t)bq0 * FF + i];
  __syncthreads();
  {
    int n0a = (wave * 2 + 0) * 16;
    int n0b = (wave * 2 + 1) * 16;
    float biasa = b2[n0a + (lane & 15)];
    float biasb = b2[n0b + (lane & 15)];
    v8f acc0, acc1;
#pragma unroll
    for (int r = 0; r < 8; ++r) { acc0[r] = biasa; acc1[r] = biasb; }
    // K tiled in 8 blocks of 256: A fragments loaded from LDS exactly once,
    // reused across both n-tiles; two independent WMMAs per A fragment.
    for (int kb = 0; kb < 8; ++kb) {
      v16bf a[8];
#pragma unroll
      for (int k8 = 0; k8 < 8; ++k8)
        a[k8] = load_a_lds(sA, lane, kb * 256 + k8 * 32, AP2);
#pragma unroll
      for (int k8 = 0; k8 < 8; ++k8) {
        int kk = kb * 256 + k8 * 32;
        v16bf bm0 = load_b_glb(w2, lane, n0a, kk, FF);
        acc0 = WMMA_BF16(a[k8], bm0, acc0);
        v16bf bm1 = load_b_glb(w2, lane, n0b, kk, FF);
        acc1 = WMMA_BF16(a[k8], bm1, acc1);
      }
    }
    int na = n0a + (lane & 15);
    int nb = n0b + (lane & 15);
    int mo = (lane & 16) ? 8 : 0;
#pragma unroll
    for (int r = 0; r < 8; ++r) {
      int m = r + mo;
      sY[m * 256 + na] = acc0[r] + (float)x1[(size_t)(bq0 + m) * C + na];
      sY[m * 256 + nb] = acc1[r] + (float)x1[(size_t)(bq0 + m) * C + nb];
    }
  }
  __syncthreads();
  ln_16x256(sY, n2_g, n2_b, red, stat);
  int b = bq0 / NQ, q0 = bq0 % NQ;
  int cc = tid;
#pragma unroll 4
  for (int m = 0; m < 16; ++m)
    out[((size_t)b * C + cc) * NQ + q0 + m] = sY[m * 256 + cc];
}

// ================================================================ host
extern "C" void kernel_launch(void* const* d_in, const int* in_sizes, int n_in,
                              void* d_out, int out_size, void* d_ws, size_t ws_size,
                              hipStream_t stream) {
  (void)in_sizes; (void)n_in; (void)out_size; (void)ws_size;
  const float* query = (const float*)d_in[0];
  const float* key = (const float*)d_in[1];
  const float* query_pos = (const float*)d_in[2];
  const float* key_pos = (const float*)d_in[3];
  const float* q_w = (const float*)d_in[4];
  const float* q_b = (const float*)d_in[5];
  const float* k_w = (const float*)d_in[6];
  const float* k_b = (const float*)d_in[7];
  const float* v_w = (const float*)d_in[8];
  const float* v_b = (const float*)d_in[9];
  const float* o_w = (const float*)d_in[10];
  const float* o_b = (const float*)d_in[11];
  const float* lin1_w = (const float*)d_in[12];
  const float* lin1_b = (const float*)d_in[13];
  const float* lin2_w = (const float*)d_in[14];
  const float* lin2_b = (const float*)d_in[15];
  const float* n1_g = (const float*)d_in[16];
  const float* n1_b = (const float*)d_in[17];
  const float* n2_g = (const float*)d_in[18];
  const float* n2_b = (const float*)d_in[19];
  const float* qpe_w = (const float*)d_in[20];
  const float* qpe_b = (const float*)d_in[21];
  const float* kpe_w = (const float*)d_in[22];
  const float* kpe_b = (const float*)d_in[23];
  float* out = (float*)d_out;

  char* ws = (char*)d_ws;
  size_t off = 0;
  auto carve = [&](size_t bytes) -> void* {
    void* p = ws + off;
    off += (bytes + 255) & ~(size_t)255;
    return p;
  };
  __bf16* keyT = (__bf16*)carve((size_t)BB * NK * C * 2);
  __bf16* wqb = (__bf16*)carve((size_t)C * C * 2);
  __bf16* wkb = (__bf16*)carve((size_t)C * C * 2);
  __bf16* wvb = (__bf16*)carve((size_t)C * C * 2);
  __bf16* wob = (__bf16*)carve((size_t)C * C * 2);
  __bf16* w1b = (__bf16*)carve((size_t)FF * C * 2);
  __bf16* w2b = (__bf16*)carve((size_t)C * FF * 2);
  int* idxb = (int*)carve((size_t)BQ * S * 4);
  unsigned* maskb = (unsigned*)carve((size_t)BQ * 4);
  __bf16* qtok = (__bf16*)carve((size_t)BQ * C * 2);
  __bf16* qin = (__bf16*)carve((size_t)BQ * C * 2);
  float* gxyz = (float*)carve((size_t)BQ * S * 3 * 4);
  __bf16* aob = (__bf16*)carve((size_t)BQ * C * 2);
  __bf16* x1b = (__bf16*)carve((size_t)BQ * C * 2);
  __bf16* hbuf = (__bf16*)carve((size_t)BQ * FF * 2);

  // 1. weights -> bf16
  cvt_bf16_kernel<<<(C * C + 255) / 256, 256, 0, stream>>>(q_w, wqb, C * C);
  cvt_bf16_kernel<<<(C * C + 255) / 256, 256, 0, stream>>>(k_w, wkb, C * C);
  cvt_bf16_kernel<<<(C * C + 255) / 256, 256, 0, stream>>>(v_w, wvb, C * C);
  cvt_bf16_kernel<<<(C * C + 255) / 256, 256, 0, stream>>>(o_w, wob, C * C);
  cvt_bf16_kernel<<<(FF * C + 255) / 256, 256, 0, stream>>>(lin1_w, w1b, FF * C);
  cvt_bf16_kernel<<<(C * FF + 255) / 256, 256, 0, stream>>>(lin2_w, w2b, C * FF);

  // 2. key transpose to (B, NK, C) bf16
  transpose_key_kernel<<<dim3(NK / 32, C / 32, BB), dim3(32, 8), 0, stream>>>(key, keyT);

  // 3. box query
  box_query_kernel<<<BB * (NQ / 256), 256, 0, stream>>>(query_pos, key_pos, idxb, maskb);

  // 4. qtok / q_in / gxyz
  build_kernel<<<BQ, 256, 0, stream>>>(query, query_pos, key_pos, qpe_w, qpe_b, idxb,
                                       qtok, qin, gxyz);

  // 5. fused gather + kpe + QKV + attention
  size_t attn_smem = (size_t)(256 * KVP * 2) * 2 + (size_t)16 * KVP * 2 +
                     (size_t)16 * 256 * 4 + (size_t)16 * S * 3 * 4 + 256 * 4;  // 299264
  attn_kernel<<<BQ / 16, 256, attn_smem, stream>>>(keyT, qin, gxyz, idxb, maskb,
                                                   kpe_w, kpe_b, wqb, q_b, wkb, k_b,
                                                   wvb, v_b, aob);

  // 6. o-proj + residual + LN1
  size_t op_smem = (size_t)16 * KVP * 2 + (size_t)16 * 256 * 4;  // 24832
  oproj_ln1_kernel<<<BQ / 16, 256, op_smem, stream>>>(aob, wob, o_b, qtok, n1_g, n1_b, x1b);

  // 7. FFN1
  size_t f1_smem = (size_t)16 * KVP * 2;  // 8448
  ffn1_kernel<<<dim3(BQ / 16, FF / 256), 256, f1_smem, stream>>>(x1b, w1b, lin1_b, hbuf);

  // 8. FFN2 + residual + LN2 + transposed output
  size_t f2_smem = (size_t)16 * AP2 * 2 + (size_t)16 * 256 * 4;  // 82176
  ffn2_ln2_out_kernel<<<BQ / 16, 256, f2_smem, stream>>>(hbuf, w2b, lin2_b, x1b,
                                                         n2_g, n2_b, out);
}
